// RGCNConvQuant_65687229825999
// MI455X (gfx1250) — 8-bit matrix operand order repaired and checked by reading, compiled, not run on hardware
//
#include <hip/hip_runtime.h>

typedef int v8i __attribute__((ext_vector_type(8)));

#define DF 128

__device__ __forceinline__ float scale_of(unsigned bits) {
  return fmaxf(__uint_as_float(bits) / 127.0f, 1e-12f);
}

__global__ __launch_bounds__(256)
void absmax_kernel(const float* __restrict__ x, long n, unsigned* __restrict__ out) {
  float m = 0.0f;
  long i = (long)blockIdx.x * blockDim.x + threadIdx.x;
  long stride = (long)gridDim.x * blockDim.x;
  for (; i < n; i += stride) m = fmaxf(m, fabsf(x[i]));
#pragma unroll
  for (int off = 16; off > 0; off >>= 1) m = fmaxf(m, __shfl_xor(m, off, 32));
  __shared__ float sm[8];
  if ((threadIdx.x & 31) == 0) sm[threadIdx.x >> 5] = m;
  __syncthreads();
  if (threadIdx.x < 8) {
    float v = sm[threadIdx.x];
#pragma unroll
    for (int off = 4; off > 0; off >>= 1) v = fmaxf(v, __shfl_xor(v, off, 32));
    if (threadIdx.x == 0) atomicMax(out, __float_as_uint(v));
  }
}

__global__ __launch_bounds__(256)
void scatter_add_kernel(const float* __restrict__ x, const int* __restrict__ src,
                        const int* __restrict__ dst, float* __restrict__ agg,
                        float* __restrict__ cnt, int E) {
  long idx = (long)blockIdx.x * blockDim.x + threadIdx.x;
  int e = (int)(idx >> 5);
  int lane = (int)(idx & 31);
  if (e >= E) return;
  int s = src[e];
  int d = dst[e];
  const float4* xp = (const float4*)(x + (long)s * DF);
  float* ap = agg + (long)d * DF + 4 * lane;
  float4 v = xp[lane];
  atomicAdd(&ap[0], v.x);
  atomicAdd(&ap[1], v.y);
  atomicAdd(&ap[2], v.z);
  atomicAdd(&ap[3], v.w);
  if (lane == 0) atomicAdd(&cnt[d], 1.0f);
}

__global__ __launch_bounds__(256)
void finalize_agg_kernel(float* __restrict__ agg, const float* __restrict__ cnt,
                         long total4, unsigned* __restrict__ out) {
  float m = 0.0f;
  long i = (long)blockIdx.x * blockDim.x + threadIdx.x;
  long stride = (long)gridDim.x * blockDim.x;
  float4* av = (float4*)agg;
  for (; i < total4; i += stride) {
    float rc = 1.0f / fmaxf(cnt[i >> 5], 1.0f);
    float4 v = av[i];
    v.x *= rc; v.y *= rc; v.z *= rc; v.w *= rc;
    av[i] = v;
    m = fmaxf(m, fmaxf(fmaxf(fabsf(v.x), fabsf(v.y)), fmaxf(fabsf(v.z), fabsf(v.w))));
  }
#pragma unroll
  for (int off = 16; off > 0; off >>= 1) m = fmaxf(m, __shfl_xor(m, off, 32));
  __shared__ float sm[8];
  if ((threadIdx.x & 31) == 0) sm[threadIdx.x >> 5] = m;
  __syncthreads();
  if (threadIdx.x < 8) {
    float v = sm[threadIdx.x];
#pragma unroll
    for (int off = 4; off > 0; off >>= 1) v = fmaxf(v, __shfl_xor(v, off, 32));
    if (threadIdx.x == 0) atomicMax(out, __float_as_uint(v));
  }
}

__global__ __launch_bounds__(256)
void quantize_kernel(const float* __restrict__ x, signed char* __restrict__ q,
                     long n4, const unsigned* __restrict__ amax) {
  float inv = 1.0f / scale_of(*amax);
  long i = (long)blockIdx.x * blockDim.x + threadIdx.x;
  long stride = (long)gridDim.x * blockDim.x;
  const float4* xv = (const float4*)x;
  unsigned* qv = (unsigned*)q;
  for (; i < n4; i += stride) {
    float4 v = xv[i];
    int q0 = (int)fminf(fmaxf(rintf(v.x * inv), -127.0f), 127.0f);
    int q1 = (int)fminf(fmaxf(rintf(v.y * inv), -127.0f), 127.0f);
    int q2 = (int)fminf(fmaxf(rintf(v.z * inv), -127.0f), 127.0f);
    int q3 = (int)fminf(fmaxf(rintf(v.w * inv), -127.0f), 127.0f);
    qv[i] = (unsigned)(q0 & 0xff) | ((unsigned)(q1 & 0xff) << 8) |
            ((unsigned)(q2 & 0xff) << 16) | ((unsigned)(q3 & 0xff) << 24);
  }
}

__global__ void prep_scales_kernel(const unsigned* __restrict__ scales,
                                   float* __restrict__ sprod) {
  if (threadIdx.x == 0 && blockIdx.x == 0) {
    sprod[0] = scale_of(scales[0]) * scale_of(scales[2]);
    sprod[1] = scale_of(scales[7]) * scale_of(scales[5]);
    sprod[2] = scale_of(scales[1]) * scale_of(scales[3]);
    sprod[3] = scale_of(scales[6]) * scale_of(scales[4]);
  }
}

__device__ __forceinline__ v8i load_A64(const signed char* rowbase, int kb, int half) {
  const signed char* p = rowbase + kb * 64;
  v8i a;
#pragma unroll
  for (int pr = 0; pr < 4; ++pr) {
    int2 v = *(const int2*)(p + (pr >> 1) * 32 + half * 16 + (pr & 1) * 8);
    a[2 * pr] = v.x;
    a[2 * pr + 1] = v.y;
  }
  return a;
}

__device__ __forceinline__ v8i load_B64(const signed char* wrow, int kb, int half) {
  int4 lo = *(const int4*)(wrow + kb * 64 + half * 16);
  int4 hi = *(const int4*)(wrow + kb * 64 + 32 + half * 16);
  v8i b;
  b[0] = lo.x; b[1] = lo.y; b[2] = lo.z; b[3] = lo.w;
  b[4] = hi.x; b[5] = hi.y; b[6] = hi.z; b[7] = hi.w;
  return b;
}

__global__ __launch_bounds__(256)
void gemm_dual_i8_kernel(const signed char* __restrict__ qA1, const signed char* __restrict__ qW1,
                         const signed char* __restrict__ qA2, const signed char* __restrict__ qW2,
                         const float* __restrict__ sprod, int i1, int i2,
                         float* __restrict__ out, int nrows) {
  int wave = threadIdx.x >> 5;
  int lane = threadIdx.x & 31;
  int half = lane >> 4;
  int m = lane & 15;
  int row0 = blockIdx.x << 4;
  int col0 = wave << 4;

  int rowA = row0 + m;
  if (rowA >= nrows) rowA = nrows - 1;

  const signed char* a1 = qA1 + (long)rowA * DF;
  const signed char* a2 = qA2 + (long)rowA * DF;
  const signed char* w1 = qW1 + (long)(col0 + m) * DF;
  const signed char* w2 = qW2 + (long)(col0 + m) * DF;

  v8i A10 = load_A64(a1, 0, half);
  v8i A11 = load_A64(a1, 1, half);
  v8i B10 = load_B64(w1, 0, half);
  v8i B11 = load_B64(w1, 1, half);
  v8i acc1 = {};
  acc1 = __builtin_amdgcn_wmma_i32_16x16x64_iu8(true, A10, true, B10, acc1, false, false);
  acc1 = __builtin_amdgcn_wmma_i32_16x16x64_iu8(true, A11, true, B11, acc1, false, false);

  v8i A20 = load_A64(a2, 0, half);
  v8i A21 = load_A64(a2, 1, half);
  v8i B20 = load_B64(w2, 0, half);
  v8i B21 = load_B64(w2, 1, half);
  v8i acc2 = {};
  acc2 = __builtin_amdgcn_wmma_i32_16x16x64_iu8(true, A20, true, B20, acc2, false, false);
  acc2 = __builtin_amdgcn_wmma_i32_16x16x64_iu8(true, A21, true, B21, acc2, false, false);

  float s1 = sprod[i1];
  float s2 = sprod[i2];

#pragma unroll
  for (int r = 0; r < 8; ++r) {
    int row = row0 + r + half * 8;
    if (row < nrows)
      out[(long)row * DF + col0 + m] = s1 * (float)acc1[r] + s2 * (float)acc2[r];
  }
}

extern "C" void kernel_launch(void* const* d_in, const int* in_sizes, int n_in,
                              void* d_out, int out_size, void* d_ws, size_t ws_size,
                              hipStream_t stream) {
  const float* x_a = (const float*)d_in[0];
  const float* x_b = (const float*)d_in[1];
  const float* w_root_a = (const float*)d_in[2];
  const float* w_root_b = (const float*)d_in[3];
  const float* w_rel1 = (const float*)d_in[4];
  const float* w_rel2 = (const float*)d_in[5];
  const int* e1s = (const int*)d_in[6];
  const int* e1d = (const int*)d_in[7];
  const int* e2s = (const int*)d_in[8];
  const int* e2d = (const int*)d_in[9];

  const long nA = (long)in_sizes[0] / DF;
  const long nB = (long)in_sizes[1] / DF;
  const int E1 = in_sizes[6];
  const int E2 = in_sizes[8];
  const long nW = (long)DF * DF;

  char* ws = (char*)d_ws;
  size_t off = 0;
  auto walloc = [&](size_t bytes) -> char* {
    char* p = ws + off;
    off = (off + bytes + 255) & ~(size_t)255;
    return p;
  };
  unsigned* scales = (unsigned*)walloc(16 * sizeof(unsigned));
  float* sprod = (float*)(scales + 8);
  float* cnt1 = (float*)walloc(nB * sizeof(float));
  float* cnt2 = (float*)walloc(nA * sizeof(float));
  float* agg1 = (float*)walloc(nB * DF * sizeof(float));
  float* agg2 = (float*)walloc(nA * DF * sizeof(float));
  size_t zero_bytes = off;
  signed char* qxa = (signed char*)walloc(nA * DF);
  signed char* qxb = (signed char*)walloc(nB * DF);
  signed char* qagg1 = (signed char*)walloc(nB * DF);
  signed char* qagg2 = (signed char*)walloc(nA * DF);
  signed char* qwra = (signed char*)walloc(nW);
  signed char* qwrb = (signed char*)walloc(nW);
  signed char* qwr1 = (signed char*)walloc(nW);
  signed char* qwr2 = (signed char*)walloc(nW);

  hipMemsetAsync(d_ws, 0, zero_bytes, stream);

  auto blocks_for = [](long n) -> int {
    long b = (n + 255) / 256;
    return (int)(b > 2048 ? 2048 : b);
  };

  absmax_kernel<<<blocks_for(nA * DF), 256, 0, stream>>>(x_a, nA * DF, scales + 0);
  absmax_kernel<<<blocks_for(nB * DF), 256, 0, stream>>>(x_b, nB * DF, scales + 1);
  absmax_kernel<<<blocks_for(nW), 256, 0, stream>>>(w_root_a, nW, scales + 2);
  absmax_kernel<<<blocks_for(nW), 256, 0, stream>>>(w_root_b, nW, scales + 3);
  absmax_kernel<<<blocks_for(nW), 256, 0, stream>>>(w_rel1, nW, scales + 4);
  absmax_kernel<<<blocks_for(nW), 256, 0, stream>>>(w_rel2, nW, scales + 5);

  scatter_add_kernel<<<(E1 + 7) / 8, 256, 0, stream>>>(x_a, e1s, e1d, agg1, cnt1, E1);
  scatter_add_kernel<<<(E2 + 7) / 8, 256, 0, stream>>>(x_b, e2s, e2d, agg2, cnt2, E2);
  finalize_agg_kernel<<<blocks_for(nB * DF / 4), 256, 0, stream>>>(agg1, cnt1, nB * DF / 4, scales + 6);
  finalize_agg_kernel<<<blocks_for(nA * DF / 4), 256, 0, stream>>>(agg2, cnt2, nA * DF / 4, scales + 7);

  quantize_kernel<<<blocks_for(nA * DF / 4), 256, 0, stream>>>(x_a, qxa, nA * DF / 4, scales + 0);
  quantize_kernel<<<blocks_for(nB * DF / 4), 256, 0, stream>>>(x_b, qxb, nB * DF / 4, scales + 1);
  quantize_kernel<<<blocks_for(nW / 4), 256, 0, stream>>>(w_root_a, qwra, nW / 4, scales + 2);
  quantize_kernel<<<blocks_for(nW / 4), 256, 0, stream>>>(w_root_b, qwrb, nW / 4, scales + 3);
  quantize_kernel<<<blocks_for(nW / 4), 256, 0, stream>>>(w_rel1, qwr1, nW / 4, scales + 4);
  quantize_kernel<<<blocks_for(nW / 4), 256, 0, stream>>>(w_rel2, qwr2, nW / 4, scales + 5);
  quantize_kernel<<<blocks_for(nB * DF / 4), 256, 0, stream>>>(agg1, qagg1, nB * DF / 4, scales + 6);
  quantize_kernel<<<blocks_for(nA * DF / 4), 256, 0, stream>>>(agg2, qagg2, nA * DF / 4, scales + 7);

  prep_scales_kernel<<<1, 32, 0, stream>>>(scales, sprod);

  float* out_a = (float*)d_out;
  float* out_b = out_a + nA * DF;
  int gridA = (int)((nA + 15) / 16);
  int gridB = (int)((nB + 15) / 16);
  gemm_dual_i8_kernel<<<gridA, 256, 0, stream>>>(qxa, qwra, qagg2, qwr2, sprod,
                                                 0, 1, out_a, (int)nA);
  gemm_dual_i8_kernel<<<gridB, 256, 0, stream>>>(qxb, qwrb, qagg1, qwr1, sprod,
                                                 2, 3, out_b, (int)nB);
}
